// GritTransformer_17806934409795
// MI455X (gfx1250) — compile-verified, matrix operand order checked by reading, not run on hardware
//
#include <hip/hip_runtime.h>
#include <cstdint>

typedef __bf16 bf16;
typedef __attribute__((ext_vector_type(16))) __bf16 v16bf;
typedef __attribute__((ext_vector_type(8)))  __bf16 v8bf;
typedef __attribute__((ext_vector_type(8)))  float  v8f;
typedef __attribute__((ext_vector_type(4)))  unsigned int v4u;
typedef __attribute__((ext_vector_type(8)))  int v8i;
typedef __attribute__((ext_vector_type(4)))  int v4i;

#define HID    128
#define NHEAD  8
#define HD     16
#define NLAYER 2
#define CLAMP  5.0f
#define BN_SC  0.99999500003749f   // 1/sqrt(1+1e-5)
#define BLK_N  64                  // N columns staged in LDS per block
#define MAX_K  256

union BFrag { v16bf v; v8bf h[2]; };

// ---------------------------------------------------------------------------
// Block-cooperative WMMA GEMM:
//   out[M x Nc] = epilogue( A[M x K](bf16) @ Bt[Nc x K]^T (bf16) )
// Bt is the weight stored transposed, so the BLK_N x K slice a block needs is
// CONTIGUOUS -> staged into LDS once per block by the Tensor Data Mover
// (tensor_load_to_lds, TENSORcnt) when available, else cooperative copy.
// Block = 8 waves; each wave owns a 16(M) x 64(N) strip = 4 WMMA accumulators,
// reusing its A fragment 4x per K-step and reading B via ds_load_b128.
// mode 0: acc + bias ; mode 1: (res + acc + bias)*scale ; mode 2: relu(acc+bias)
// ---------------------------------------------------------------------------
__global__ void __launch_bounds__(256)
gemm_bf16_wmma(const bf16* __restrict__ A, const bf16* __restrict__ Bt,
               const float* __restrict__ bias, const float* __restrict__ res,
               float* __restrict__ outF, bf16* __restrict__ outB,
               int M, int K, int Nc, int mode, float scale)
{
    __shared__ bf16 Bs[BLK_N * MAX_K];          // 32 KB max
    const int tid  = threadIdx.x;
    const int lane = tid & 31;
    const int wave = tid >> 5;
    const int nBlocks = Nc / BLK_N;
    const int mBlk = (int)blockIdx.x / nBlocks;
    const int nBlk = (int)blockIdx.x % nBlocks;

    // ---- stage B slice (BLK_N rows x K cols of Bt, contiguous) into LDS ----
    const bf16* Bsrc = Bt + (long long)nBlk * BLK_N * K;
    const int nElem = BLK_N * K;
#if defined(__has_builtin) && __has_builtin(__builtin_amdgcn_tensor_load_to_lds)
    if (wave == 0) {
        unsigned long long ga = (unsigned long long)(uintptr_t)Bsrc;
        unsigned int la = (unsigned int)(size_t)Bs;          // LDS byte offset
        v4u g0;
        g0[0] = 1u;                                          // count=1 (valid)
        g0[1] = la;                                          // lds_addr [63:32]
        g0[2] = (unsigned int)(ga & 0xffffffffu);            // global_addr lo
        g0[3] = (unsigned int)((ga >> 32) & 0x01ffffffu)     // global_addr [120:64]
              | (2u << 30);                                  // type=2 (image)
        v8i g1;
        g1[0] = (int)(1u << 16);                             // data_size=2 bytes
        g1[1] = (int)(((unsigned)K & 0xffffu) << 16);        // tensor_dim0 lo16
        g1[2] = (int)(((unsigned)BLK_N & 0xffffu) << 16);    // dim0 hi=0 | tensor_dim1 lo16
        g1[3] = (int)((unsigned)K << 16);                    // dim1 hi=0 | tile_dim0=K
        g1[4] = BLK_N;                                       // tile_dim1 | tile_dim2=0
        g1[5] = K;                                           // tensor_dim0_stride lo32
        g1[6] = 0;
        g1[7] = 0;
        v4i z = {0, 0, 0, 0};
#if __clang_major__ >= 23
        v8i z8 = {0, 0, 0, 0, 0, 0, 0, 0};
        __builtin_amdgcn_tensor_load_to_lds(g0, g1, z, z, z8, 0);
#else
        __builtin_amdgcn_tensor_load_to_lds(g0, g1, z, z, 0);
#endif
        __builtin_amdgcn_s_wait_tensorcnt(0);
    }
#else
    for (int c = tid * 8; c < nElem; c += 256 * 8)
        *(v8bf*)(Bs + c) = *(const v8bf*)(Bsrc + c);
#endif
    __syncthreads();

    const int hl  = lane >> 4;      // K-half held by this lane
    const int r16 = lane & 15;      // A: row in tile ; C/D: col in tile
    const int mBase = mBlk * 128 + wave * 16;
    if (mBase >= M) return;

    v8f acc[4] = {{}, {}, {}, {}};
    const bf16* Arow = A + (long long)(mBase + r16) * K + hl * 8;
    for (int kb = 0; kb < K; kb += 32) {
        BFrag a;
        a.h[0] = *(const v8bf*)(Arow + kb);
        a.h[1] = *(const v8bf*)(Arow + kb + 16);
#pragma unroll
        for (int j = 0; j < 4; ++j) {
            BFrag b;
            const bf16* brow = Bs + (j * 16 + r16) * K + kb + hl * 8;
            b.h[0] = *(const v8bf*)(brow);
            b.h[1] = *(const v8bf*)(brow + 16);
            acc[j] = __builtin_amdgcn_wmma_f32_16x16x32_bf16(
                         false, a.v, false, b.v, (short)0, acc[j], false, false);
        }
    }

#pragma unroll
    for (int j = 0; j < 4; ++j) {
        const int n = nBlk * BLK_N + j * 16 + r16;
        const float bv = bias ? bias[n] : 0.0f;
#pragma unroll
        for (int r = 0; r < 8; ++r) {
            const int m = mBase + r + hl * 8;
            float v = acc[j][r] + bv;
            const long long o = (long long)m * Nc + n;
            if (mode == 1)      v = (res[o] + v) * scale;
            else if (mode == 2) v = v > 0.f ? v : 0.f;
            if (outF) outF[o] = v;
            if (outB) outB[o] = (bf16)v;
        }
    }
}

// ---------------------------------------------------------------------------
// Small helper kernels
// ---------------------------------------------------------------------------
__global__ void cvt_bf16(const float* __restrict__ in, bf16* __restrict__ ob,
                         float* __restrict__ of, long long n)
{
    long long i = (long long)blockIdx.x * blockDim.x + threadIdx.x;
    if (i >= n) return;
    float v = in[i];
    ob[i] = (bf16)v;
    if (of) of[i] = v;
}

// Wt[n*K + k] = W[k*Nc + n]   (bf16)
__global__ void wt_bf16(const float* __restrict__ W, bf16* __restrict__ Wt,
                        int K, int Nc)
{
    int i = blockIdx.x * blockDim.x + threadIdx.x;
    if (i >= K * Nc) return;
    int k = i / Nc, n = i % Nc;
    Wt[(long long)n * K + k] = (bf16)W[i];
}

__global__ void deg_accum(const int* __restrict__ dst, float* __restrict__ deg,
                          long long E)
{
    long long i = (long long)blockIdx.x * blockDim.x + threadIdx.x;
    if (i >= E) return;
    atomicAdd(&deg[dst[i]], 1.0f);
}

__global__ void deg_log(const float* __restrict__ deg, float* __restrict__ ldg,
                        int N)
{
    int i = blockIdx.x * blockDim.x + threadIdx.x;
    if (i >= N) return;
    ldg[i] = logf(deg[i] + 1.0f);
}

// ---------------------------------------------------------------------------
// Edge pass 1: e_t, attention logit (clamped -> exp without segment_max),
// atomic denominator accumulation.  One thread per (edge, head).
// ---------------------------------------------------------------------------
__global__ void edge_pass1(const int* __restrict__ src, const int* __restrict__ dst,
                           const float* __restrict__ Qh, const float* __restrict__ Kh,
                           const float* __restrict__ Ep, const float* __restrict__ Aw,
                           bf16* __restrict__ et_b, float* __restrict__ a_sc,
                           float* __restrict__ denom, long long E)
{
    long long idx = (long long)blockIdx.x * blockDim.x + threadIdx.x;
    if (idx >= E * NHEAD) return;
    const long long e = idx >> 3;
    const int h = (int)(idx & 7);
    const int s = src[e], d = dst[e];
    const float* kp = Kh + (long long)s * HID + h * HD;
    const float* qp = Qh + (long long)d * HID + h * HD;
    const float* ew = Ep + e * (2 * HID) + h * (2 * HD);
    const float* eb = ew + HD;
    float sc = 0.f;
    float et[HD];
#pragma unroll
    for (int c = 0; c < HD; ++c) {
        float sv = (kp[c] + qp[c]) * ew[c];
        float ss = sv > 0.f ? sqrtf(sv) : (sv < 0.f ? -sqrtf(-sv) : 0.f);
        float t  = ss + eb[c];
        t = t > 0.f ? t : 0.f;           // relu -> e_t
        et[c] = t;
        sc += t * Aw[c * NHEAD + h];
    }
    sc = fminf(CLAMP, fmaxf(-CLAMP, sc));
    float a = expf(sc);                  // clamped => no segment_max needed
#pragma unroll
    for (int c = 0; c < HD; ++c)
        et_b[e * HID + h * HD + c] = (bf16)et[c];
    a_sc[idx] = a;
    atomicAdd(&denom[(long long)d * NHEAD + h], a);
}

// ---------------------------------------------------------------------------
// Edge pass 2: attn = a/denom ; atomic scatter of attn*Vh and attn*e_t.
// ---------------------------------------------------------------------------
__global__ void edge_pass2(const int* __restrict__ src, const int* __restrict__ dst,
                           const float* __restrict__ Vh, const bf16* __restrict__ et_b,
                           const float* __restrict__ a_sc, const float* __restrict__ denom,
                           float* __restrict__ wV, float* __restrict__ rowV, long long E)
{
    long long idx = (long long)blockIdx.x * blockDim.x + threadIdx.x;
    if (idx >= E * NHEAD) return;
    const long long e = idx >> 3;
    const int h = (int)(idx & 7);
    const int s = src[e], d = dst[e];
    const float attn = a_sc[idx] / (denom[(long long)d * NHEAD + h] + 1e-16f);
    const float* vp = Vh + (long long)s * HID + h * HD;
    const bf16* ep  = et_b + e * HID + h * HD;
    float* wp = wV   + (long long)d * HID + h * HD;
    float* rp = rowV + (long long)d * HID + h * HD;
#pragma unroll
    for (int c = 0; c < HD; ++c) {
        atomicAdd(&wp[c], vp[c] * attn);
        atomicAdd(&rp[c], (float)ep[c] * attn);
    }
}

// ---------------------------------------------------------------------------
// Node combine: wV += rowV @ VeRow ; degree scaling ; emit bf16 h_pre.
// ---------------------------------------------------------------------------
__global__ void node_combine(const float* __restrict__ wV, const float* __restrict__ rowV,
                             const float* __restrict__ VeRow, const float* __restrict__ dcoef,
                             const float* __restrict__ ldg, bf16* __restrict__ hpre_b, int N)
{
    long long idx = (long long)blockIdx.x * blockDim.x + threadIdx.x;
    if (idx >= (long long)N * NHEAD) return;
    const int n = (int)(idx >> 3);
    const int h = (int)(idx & 7);
    float rv[HD];
#pragma unroll
    for (int d = 0; d < HD; ++d) rv[d] = rowV[(long long)n * HID + h * HD + d];
    const float ld = ldg[n];
#pragma unroll
    for (int c = 0; c < HD; ++c) {
        float acc = wV[(long long)n * HID + h * HD + c];
#pragma unroll
        for (int d = 0; d < HD; ++d)
            acc += rv[d] * VeRow[d * (NHEAD * HD) + h * HD + c];
        const int col = h * HD + c;
        float hv = acc * dcoef[col * 2 + 0] + acc * ld * dcoef[col * 2 + 1];
        hpre_b[(long long)n * HID + col] = (bf16)hv;
    }
}

// ---------------------------------------------------------------------------
extern "C" void kernel_launch(void* const* d_in, const int* in_sizes, int n_in,
                              void* d_out, int out_size, void* d_ws, size_t ws_size,
                              hipStream_t stream)
{
    const float* x_in  = (const float*)d_in[0];
    const int*   ei    = (const int*)  d_in[1];
    const float* ea_in = (const float*)d_in[2];
    const float* Wq    = (const float*)d_in[3];
    const float* bq    = (const float*)d_in[4];
    const float* Wk    = (const float*)d_in[5];
    const float* Wv    = (const float*)d_in[6];
    const float* We    = (const float*)d_in[7];
    const float* be    = (const float*)d_in[8];
    const float* Aw    = (const float*)d_in[9];
    const float* VeRow = (const float*)d_in[10];
    const float* Woh   = (const float*)d_in[11];
    const float* boh   = (const float*)d_in[12];
    const float* Woe   = (const float*)d_in[13];
    const float* boe   = (const float*)d_in[14];
    const float* dcoef = (const float*)d_in[15];
    const float* W1    = (const float*)d_in[16];
    const float* b1    = (const float*)d_in[17];
    const float* W2    = (const float*)d_in[18];
    const float* b2    = (const float*)d_in[19];

    const long long N = in_sizes[0] / HID;
    const long long E = in_sizes[1] / 2;
    const int* srcI = ei;
    const int* dstI = ei + E;

    // ---- workspace carve ----
    char* p = (char*)d_ws;
    auto carve = [&](size_t bytes) -> char* {
        char* r = p; p += (bytes + 255) & ~(size_t)255; return r;
    };
    float* xf   = (float*)carve(N * HID * 4);
    bf16*  xb   = (bf16*) carve(N * HID * 2);
    float* ef   = (float*)carve(E * HID * 4);
    bf16*  eb   = (bf16*) carve(E * HID * 2);
    float* Qh   = (float*)carve(N * HID * 4);
    float* Khm  = (float*)carve(N * HID * 4);
    float* Vh   = (float*)carve(N * HID * 4);
    float* Ep   = (float*)carve(E * 2 * HID * 4);
    bf16*  et_b = (bf16*) carve(E * HID * 2);
    float* a_sc = (float*)carve(E * NHEAD * 4);
    float* den  = (float*)carve(N * NHEAD * 4);
    float* wV   = (float*)carve(N * HID * 4);
    float* rowV = (float*)carve(N * HID * 4);
    bf16*  hpre = (bf16*) carve(N * HID * 2);
    float* h_f  = (float*)carve(N * HID * 4);
    bf16*  h_b  = (bf16*) carve(N * HID * 2);
    float* t_f  = (float*)carve(N * 2 * HID * 4);
    bf16*  t_b  = (bf16*) carve(N * 2 * HID * 2);
    float* deg  = (float*)carve(N * 4);
    float* ldg  = (float*)carve(N * 4);
    bf16* WqT  = (bf16*)carve(NLAYER * HID * HID * 2);
    bf16* WkT  = (bf16*)carve(NLAYER * HID * HID * 2);
    bf16* WvT  = (bf16*)carve(NLAYER * HID * HID * 2);
    bf16* WeT  = (bf16*)carve(NLAYER * HID * 2 * HID * 2);
    bf16* WohT = (bf16*)carve(NLAYER * HID * HID * 2);
    bf16* WoeT = (bf16*)carve(NLAYER * HID * HID * 2);
    bf16* W1T  = (bf16*)carve(NLAYER * HID * 2 * HID * 2);
    bf16* W2T  = (bf16*)carve(NLAYER * 2 * HID * HID * 2);
    if ((size_t)(p - (char*)d_ws) > ws_size) return;

    const int TB = 256;
    auto blocks = [&](long long n) { return (unsigned)((n + TB - 1) / TB); };

    // ---- weight transpose + bf16 quantize ----
    for (int l = 0; l < NLAYER; ++l) {
        const int sq = HID * HID, dq = HID * 2 * HID;
        wt_bf16<<<blocks(sq), TB, 0, stream>>>(Wq  + (long long)l * sq, WqT  + (long long)l * sq, HID, HID);
        wt_bf16<<<blocks(sq), TB, 0, stream>>>(Wk  + (long long)l * sq, WkT  + (long long)l * sq, HID, HID);
        wt_bf16<<<blocks(sq), TB, 0, stream>>>(Wv  + (long long)l * sq, WvT  + (long long)l * sq, HID, HID);
        wt_bf16<<<blocks(dq), TB, 0, stream>>>(We  + (long long)l * dq, WeT  + (long long)l * dq, HID, 2 * HID);
        wt_bf16<<<blocks(sq), TB, 0, stream>>>(Woh + (long long)l * sq, WohT + (long long)l * sq, HID, HID);
        wt_bf16<<<blocks(sq), TB, 0, stream>>>(Woe + (long long)l * sq, WoeT + (long long)l * sq, HID, HID);
        wt_bf16<<<blocks(dq), TB, 0, stream>>>(W1  + (long long)l * dq, W1T  + (long long)l * dq, HID, 2 * HID);
        wt_bf16<<<blocks(dq), TB, 0, stream>>>(W2  + (long long)l * dq, W2T  + (long long)l * dq, 2 * HID, HID);
    }

    // ---- activations -> (bf16, f32 working copy) ----
    cvt_bf16<<<blocks(N * HID), TB, 0, stream>>>(x_in,  xb, xf, N * HID);
    cvt_bf16<<<blocks(E * HID), TB, 0, stream>>>(ea_in, eb, ef, E * HID);

    // ---- degrees (once) ----
    hipMemsetAsync(deg, 0, N * 4, stream);
    deg_accum<<<blocks(E), TB, 0, stream>>>(dstI, deg, E);
    deg_log<<<blocks(N), TB, 0, stream>>>(deg, ldg, (int)N);

    auto gemm = [&](const bf16* A, const bf16* Bt, const float* bias, const float* res,
                    float* oF, bf16* oB, long long M, int K, int Nc, int mode) {
        unsigned mBlocks = (unsigned)((M + 127) / 128);
        unsigned nBlocks = (unsigned)(Nc / BLK_N);
        gemm_bf16_wmma<<<mBlocks * nBlocks, 256, 0, stream>>>(
            A, Bt, bias, res, oF, oB, (int)M, K, Nc, mode, BN_SC);
    };

    for (int l = 0; l < NLAYER; ++l) {
        const bool last = (l == NLAYER - 1);
        const long long sq = (long long)HID * HID, dq = (long long)HID * 2 * HID;

        hipMemsetAsync(den,  0, N * NHEAD * 4, stream);
        hipMemsetAsync(wV,   0, N * HID * 4, stream);
        hipMemsetAsync(rowV, 0, N * HID * 4, stream);

        // Q/K/V + edge features (WMMA)
        gemm(xb, WqT + l * sq, bq + l * HID, nullptr, Qh,  nullptr, N, HID, HID, 0);
        gemm(xb, WkT + l * sq, nullptr,      nullptr, Khm, nullptr, N, HID, HID, 0);
        gemm(xb, WvT + l * sq, nullptr,      nullptr, Vh,  nullptr, N, HID, HID, 0);
        gemm(eb, WeT + l * dq, be + l * 2 * HID, nullptr, Ep, nullptr, E, HID, 2 * HID, 0);

        // sparse attention (atomic segment softmax; logits clamped => no max pass)
        edge_pass1<<<blocks(E * NHEAD), TB, 0, stream>>>(
            srcI, dstI, Qh, Khm, Ep, Aw + (long long)l * HD * NHEAD,
            et_b, a_sc, den, E);
        edge_pass2<<<blocks(E * NHEAD), TB, 0, stream>>>(
            srcI, dstI, Vh, et_b, a_sc, den, wV, rowV, E);
        node_combine<<<blocks(N * NHEAD), TB, 0, stream>>>(
            wV, rowV, VeRow + (long long)l * HD * NHEAD * HD,
            dcoef + (long long)l * HID * 2, ldg, hpre, (int)N);

        // output projections + residual + BN scale (fused epilogues)
        gemm(hpre, WohT + l * sq, boh + l * HID, xf, h_f, h_b, N, HID, HID, 1);
        float* eOut = last ? ((float*)d_out + N * HID) : ef;
        gemm(et_b, WoeT + l * sq, boe + l * HID, ef, eOut, eb, E, HID, HID, 1);

        // FFN: relu(h@W1+b1)@W2+b2, residual + BN scale
        gemm(h_b, W1T + l * dq, b1 + l * 2 * HID, nullptr, t_f, t_b, N, HID, 2 * HID, 2);
        float* xOut = last ? (float*)d_out : xf;
        gemm(t_b, W2T + l * dq, b2 + l * HID, h_f, xOut, xb, N, 2 * HID, HID, 1);
    }
}
